// GNNnet_65901978190155
// MI455X (gfx1250) — matrix operand order repaired and checked by reading, not compiled, not run on hardware
//
#include <hip/hip_runtime.h>
#include <hip/hip_bf16.h>

// ---------------------------------------------------------------------------
// GNN forward (SAGEConv x2 + MLPs + global_add_pool) for MI455X (gfx1250).
//
// Memory-bound by edge scatter (~2.5 GB irregular traffic; agg buffers fit in
// 192MB L2). Dense linears (~3 GFLOP total) run on v_wmma_f32_16x16x32_f16
// with f32 accumulation; one wave32 computes one 16-node x 16-feature tile.
// Weights are converted to f16 once per launch into the workspace, so all
// fragment loads are unconditional vector loads (ds_load_b128 /
// global_load_b128) with no per-tile conversion work.
// ---------------------------------------------------------------------------

typedef __attribute__((ext_vector_type(16))) _Float16 v16h;
typedef __attribute__((ext_vector_type(8)))  _Float16 v8h;
typedef __attribute__((ext_vector_type(4)))  _Float16 v4h;
typedef __attribute__((ext_vector_type(8)))  float    v8f;
typedef __attribute__((ext_vector_type(4)))  float    v4f;

#define N_NODES  50000
#define N_EDGES  1600000
#define N_GRAPHS 64

// ---- WMMA fragment builders (layouts per CDNA5 ISA §7.12.2) ----
// A (16M x 32K, f16): lane L holds row m = L&15.
//   lanes 0-15 : elems 0-7 -> K k0+0..7,  elems 8-15 -> K k0+16..23
//   lanes 16-31: elems 0-7 -> K k0+8..15, elems 8-15 -> K k0+24..31
// Each half is 8 consecutive f16 = one 16B LDS load. Requires ld, k0
// multiples of 32 (zero-padded when needed) and 16B-aligned tiles.
__device__ __forceinline__ v16h load_A_frag(const _Float16* act, int ld,
                                            int k0, int lane) {
    const int m  = lane & 15;
    const int kh = (lane >> 4) << 3;               // 0 or 8
    const _Float16* base = act + m * ld + k0 + kh;
    v8h lo = *(const v8h*)(base);                  // K = k0+kh .. +7
    v8h hi = *(const v8h*)(base + 16);             // K = k0+16+kh .. +7
    return __builtin_shufflevector(lo, hi, 0, 1, 2, 3, 4, 5, 6, 7,
                                   8, 9, 10, 11, 12, 13, 14, 15);
}

// B (32K x 16N, f16) from a row-major f16 weight image W[Nout][Kdim]
// (out = act @ W^T so B[k][n] = W[n][k]): lane L holds column n = n0+(L&15),
// K order matches A: elems 0-7 -> k0+8*(L>>4)+0..7, elems 8-15 -> +16 -> two 16B loads, zero cvt.
__device__ __forceinline__ v16h load_B_frag_h(const _Float16* __restrict__ W,
                                              int Kdim, int n0, int k0,
                                              int lane) {
    const int n  = n0 + (lane & 15);
    const int kb = k0 + ((lane >> 4) << 3);        // +0 or +8
    const _Float16* base = W + n * Kdim + kb;
    v8h lo = *(const v8h*)(base);
    v8h hi = *(const v8h*)(base + 16);
    return __builtin_shufflevector(lo, hi, 0, 1, 2, 3, 4, 5, 6, 7,
                                   8, 9, 10, 11, 12, 13, 14, 15);
}

__device__ __forceinline__ v8f wmma_f16(v16h a, v16h b, v8f c) {
    return __builtin_amdgcn_wmma_f32_16x16x32_f16(false, a, false, b,
                                                  (short)0, c, false, false);
}

__device__ __forceinline__ v4h cvt4(v4f q) {
    v4h r;
#pragma unroll
    for (int j = 0; j < 4; ++j) r[j] = (_Float16)q[j];
    return r;
}

// ---------------------------------------------------------------------------
__global__ void zero_kernel(float* __restrict__ p, int n) {
    int stride = gridDim.x * blockDim.x;
    for (int i = blockIdx.x * blockDim.x + threadIdx.x; i < n; i += stride)
        p[i] = 0.0f;
}

// One-shot per-launch weight conversion f32 -> f16 (launch-invariant data).
__global__ void f32_to_f16_kernel(const float* __restrict__ src,
                                  _Float16* __restrict__ dst, int n) {
    int stride = gridDim.x * blockDim.x;
    for (int i = blockIdx.x * blockDim.x + threadIdx.x; i < n; i += stride)
        dst[i] = (_Float16)src[i];
}

// W1a [32][16] f32 -> zero-padded [32][32] f16 image.
__global__ void pad_w1a_kernel(const float* __restrict__ src,
                               _Float16* __restrict__ dst) {
    int i = blockIdx.x * blockDim.x + threadIdx.x;   // 1024 threads
    if (i < 32 * 32) {
        int n = i >> 5, k = i & 31;
        dst[i] = (k < 16) ? (_Float16)src[n * 16 + k] : (_Float16)0.0f;
    }
}

// Fused MLP1: h1 = relu(relu(x@W1a^T+b1a)@W1b^T+b1b)  (leaky after relu = id)
// One wave per 16-node tile. 16->32 (2 WMMA) -> LDS f16 -> 32->64 (4 WMMA).
__global__ void mlp1_kernel(const float* __restrict__ x,
                            const _Float16* __restrict__ hWa,  // [32][32] padded
                            const float* __restrict__ b1a,
                            const _Float16* __restrict__ hW1b, // [64][32]
                            const float* __restrict__ b1b,
                            float* __restrict__ h1) {
    __shared__ __align__(16) _Float16 sX[16][32];   // K=16 zero-padded to 32
    __shared__ __align__(16) _Float16 sT[16][32];
    const int nb = blockIdx.x * 16;
    const int lane = threadIdx.x;

    // Zero the padded tile, then fill the live region with float4 chunks.
    for (int i = lane; i < 16 * 32; i += 32) (&sX[0][0])[i] = (_Float16)0.0f;
    __syncthreads();
    for (int i = lane; i < 16 * 4; i += 32) {        // 4 chunks of 4 per row
        int m = i >> 2, kc = (i & 3) << 2;
        v4f q = *(const v4f*)&x[(nb + m) * 16 + kc];
        *(v4h*)&sX[m][kc] = cvt4(q);
    }
    __syncthreads();

    // Stage 1: [16x16] @ [16x32]^T, K padded to 32
#pragma unroll
    for (int nt = 0; nt < 2; ++nt) {
        v8f c = {};
        v16h a = load_A_frag(&sX[0][0], 32, 0, lane);
        v16h b = load_B_frag_h(hWa, 32, nt * 16, 0, lane);
        c = wmma_f16(a, b, c);
        int n = nt * 16 + (lane & 15);
        float bias = b1a[n];
#pragma unroll
        for (int r = 0; r < 8; ++r) {
            int m = r + ((lane >> 4) << 3);
            float v = c[r] + bias;
            sT[m][n] = (_Float16)(v > 0.0f ? v : 0.0f);
        }
    }
    __syncthreads();

    // Stage 2: [16x32] @ [32x64]^T, K=32 exact
#pragma unroll
    for (int nt = 0; nt < 4; ++nt) {
        v8f c = {};
        v16h a = load_A_frag(&sT[0][0], 32, 0, lane);
        v16h b = load_B_frag_h(hW1b, 32, nt * 16, 0, lane);
        c = wmma_f16(a, b, c);
        int n = nt * 16 + (lane & 15);
        float bias = b1b[n];
#pragma unroll
        for (int r = 0; r < 8; ++r) {
            int m = r + ((lane >> 4) << 3);
            float v = c[r] + bias;
            h1[(nb + m) * 64 + n] = (v > 0.0f ? v : 0.0f);
        }
    }
}

// In-degree count: one f32 atomic per edge.
__global__ void deg_kernel(const int* __restrict__ ei, float* __restrict__ deg,
                           int nEdges) {
    int e = blockIdx.x * blockDim.x + threadIdx.x;
    if (e < nEdges) atomicAdd(&deg[ei[nEdges + e]], 1.0f);
}

// Edge scatter: agg[dst] += h[src], one lane per (edge, feature).
// Coalesced gather on src row, L2-resident atomic destinations (no-return
// global_atomic_add_f32 -> STOREcnt path).
template <int F>
__global__ void scatter_kernel(const float* __restrict__ h,
                               const int* __restrict__ ei,
                               float* __restrict__ agg, int nEdges) {
    unsigned tid = blockIdx.x * blockDim.x + threadIdx.x;
    int e = (int)(tid / F);
    int f = (int)(tid % F);
    if (e >= nEdges) return;
    int s = ei[e];
    int d = ei[nEdges + e];
    atomicAdd(&agg[d * F + f], h[s * F + f]);
}

// SAGE combine: out = leaky_relu( (agg/deg) @ Wl^T + bl + act @ Wr^T )
// One wave per 16-node tile; (NOUT/16) * (K/32) * 2 WMMAs per tile.
template <int K, int NOUT>
__global__ void sage_combine_kernel(const float* __restrict__ act,
                                    const float* __restrict__ agg,
                                    const float* __restrict__ deg,
                                    const _Float16* __restrict__ Wl,  // f16 image
                                    const float* __restrict__ bl,
                                    const _Float16* __restrict__ Wr,  // f16 image
                                    float* __restrict__ out) {
    __shared__ __align__(16) _Float16 sMean[16][K];
    __shared__ __align__(16) _Float16 sRoot[16][K];
    const int nb = blockIdx.x * 16;
    const int lane = threadIdx.x;

    // Stage mean & root tiles, float4 in -> 4xf16 out per lane (ds_store_b64).
    for (int i = lane; i < (16 * K) / 4; i += 32) {
        int m = (i << 2) / K, k = (i << 2) % K;    // K pow2: shifts
        int node = nb + m;
        float d = deg[node];
        d = (d < 1.0f) ? 1.0f : d;
        float inv = 1.0f / d;
        v4f qa = *(const v4f*)&agg[node * K + k];
        v4f qr = *(const v4f*)&act[node * K + k];
#pragma unroll
        for (int j = 0; j < 4; ++j) qa[j] *= inv;
        *(v4h*)&sMean[m][k] = cvt4(qa);
        *(v4h*)&sRoot[m][k] = cvt4(qr);
    }
    __syncthreads();

#pragma unroll
    for (int nt = 0; nt < NOUT / 16; ++nt) {
        v8f c = {};
        const int n0 = nt * 16;
#pragma unroll
        for (int k0 = 0; k0 < K; k0 += 32) {
            v16h a = load_A_frag(&sMean[0][0], K, k0, lane);
            v16h b = load_B_frag_h(Wl, K, n0, k0, lane);
            c = wmma_f16(a, b, c);
            a = load_A_frag(&sRoot[0][0], K, k0, lane);
            b = load_B_frag_h(Wr, K, n0, k0, lane);
            c = wmma_f16(a, b, c);
        }
        int n = n0 + (lane & 15);
        float bias = bl[n];
#pragma unroll
        for (int r = 0; r < 8; ++r) {
            int m = r + ((lane >> 4) << 3);
            float v = c[r] + bias;
            out[(nb + m) * NOUT + n] = (v > 0.0f) ? v : 0.01f * v;
        }
    }
}

// Fused MLP2 (64->32->10, relu/relu) + global_add_pool via atomics.
__global__ void mlp2_pool_kernel(const float* __restrict__ h3,
                                 const int* __restrict__ batch,
                                 const float* __restrict__ W2a, const float* __restrict__ b2a,
                                 const float* __restrict__ W2b, const float* __restrict__ b2b,
                                 float* __restrict__ out, int nNodes) {
    __shared__ float sWa[32 * 64];
    __shared__ float sba[32];
    __shared__ float sWb[10 * 32];
    __shared__ float sbb[10];
    for (int i = threadIdx.x; i < 32 * 64; i += blockDim.x) sWa[i] = W2a[i];
    for (int i = threadIdx.x; i < 32;      i += blockDim.x) sba[i] = b2a[i];
    for (int i = threadIdx.x; i < 10 * 32; i += blockDim.x) sWb[i] = W2b[i];
    for (int i = threadIdx.x; i < 10;      i += blockDim.x) sbb[i] = b2b[i];
    __syncthreads();

    int n = blockIdx.x * blockDim.x + threadIdx.x;
    if (n >= nNodes) return;

    float r[64];
#pragma unroll
    for (int k = 0; k < 64; ++k) r[k] = h3[n * 64 + k];

    float t[32];
#pragma unroll
    for (int o = 0; o < 32; ++o) {
        float s = sba[o];
#pragma unroll
        for (int k = 0; k < 64; ++k) s += r[k] * sWa[o * 64 + k];
        t[o] = (s > 0.0f) ? s : 0.0f;
    }
    int g = batch[n];
#pragma unroll
    for (int o = 0; o < 10; ++o) {
        float s = sbb[o];
#pragma unroll
        for (int k = 0; k < 32; ++k) s += t[k] * sWb[o * 32 + k];
        s = (s > 0.0f) ? s : 0.0f;
        atomicAdd(&out[g * 10 + o], s);
    }
}

// ---------------------------------------------------------------------------
extern "C" void kernel_launch(void* const* d_in, const int* in_sizes, int n_in,
                              void* d_out, int out_size, void* d_ws, size_t ws_size,
                              hipStream_t stream) {
    (void)in_sizes; (void)n_in; (void)out_size; (void)ws_size;

    const float* x    = (const float*)d_in[0];
    const int*   ei   = (const int*)d_in[1];     // [2, N_EDGES] int32 (jax x64 off)
    /* d_in[2] = edge_weight: unused by reference SAGE mean aggregation */
    const int*   batch= (const int*)d_in[3];
    const float* W1a  = (const float*)d_in[4];
    const float* b1a  = (const float*)d_in[5];
    const float* W1b  = (const float*)d_in[6];
    const float* b1b  = (const float*)d_in[7];
    const float* Wl1  = (const float*)d_in[8];
    const float* bl1  = (const float*)d_in[9];
    const float* Wr1  = (const float*)d_in[10];
    const float* Wl2  = (const float*)d_in[11];
    const float* bl2  = (const float*)d_in[12];
    const float* Wr2  = (const float*)d_in[13];
    const float* W2a  = (const float*)d_in[14];
    const float* b2a  = (const float*)d_in[15];
    const float* W2b  = (const float*)d_in[16];
    const float* b2b  = (const float*)d_in[17];

    float* out = (float*)d_out;
    float* ws  = (float*)d_ws;

    // Workspace layout (f32 elements); zeroed region first & contiguous.
    float* agg1 = ws;                         // 50000*64  = 3,200,000
    float* agg2 = agg1 + 3200000;             // 50000*128 = 6,400,000
    float* deg  = agg2 + 6400000;             // 50,000
    float* h1   = deg  + 50000;               // 3,200,000
    float* h2   = h1   + 3200000;             // 6,400,000
    float* h3   = h2   + 6400000;             // 3,200,000
    // f16 weight images (byte offset 89,800,000, 16B aligned):
    _Float16* hf   = (_Float16*)(h3 + 3200000);
    _Float16* hWa  = hf;                      // [32][32] padded  (1024)
    _Float16* hW1b = hWa  + 1024;             // [64][32]         (2048)
    _Float16* hWl1 = hW1b + 2048;             // [128][64]        (8192)
    _Float16* hWr1 = hWl1 + 8192;             // [128][64]        (8192)
    _Float16* hWl2 = hWr1 + 8192;             // [64][128]        (8192)
    _Float16* hWr2 = hWl2 + 8192;             // [64][128]        (8192)
                                              // total ~89.9 MB

    zero_kernel<<<2048, 256, 0, stream>>>(agg1, 3200000 + 6400000 + 50000);
    zero_kernel<<<4, 256, 0, stream>>>(out, N_GRAPHS * 10);

    // Per-launch weight conversion (tiny; keeps inner loops conversion-free).
    pad_w1a_kernel<<<4, 256, 0, stream>>>(W1a, hWa);
    f32_to_f16_kernel<<<8,  256, 0, stream>>>(W1b, hW1b, 2048);
    f32_to_f16_kernel<<<32, 256, 0, stream>>>(Wl1, hWl1, 8192);
    f32_to_f16_kernel<<<32, 256, 0, stream>>>(Wr1, hWr1, 8192);
    f32_to_f16_kernel<<<32, 256, 0, stream>>>(Wl2, hWl2, 8192);
    f32_to_f16_kernel<<<32, 256, 0, stream>>>(Wr2, hWr2, 8192);

    mlp1_kernel<<<N_NODES / 16, 32, 0, stream>>>(x, hWa, b1a, hW1b, b1b, h1);

    deg_kernel<<<(N_EDGES + 255) / 256, 256, 0, stream>>>(ei, deg, N_EDGES);

    scatter_kernel<64><<<(N_EDGES * 64u) / 256, 256, 0, stream>>>(h1, ei, agg1, N_EDGES);
    sage_combine_kernel<64, 128><<<N_NODES / 16, 32, 0, stream>>>(
        h1, agg1, deg, hWl1, bl1, hWr1, h2);

    scatter_kernel<128><<<(N_EDGES * 128u) / 256, 256, 0, stream>>>(h2, ei, agg2, N_EDGES);
    sage_combine_kernel<128, 64><<<N_NODES / 16, 32, 0, stream>>>(
        h2, agg2, deg, hWl2, bl2, hWr2, h3);

    mlp2_pool_kernel<<<(N_NODES + 255) / 256, 256, 0, stream>>>(
        h3, batch, W2a, b2a, W2b, b2b, out, N_NODES);
}
